// SAE_59373627899942
// MI455X (gfx1250) — matrix operand order repaired and checked by reading, not compiled, not run on hardware
//
#include <hip/hip_runtime.h>

#define TOK   4096
#define ACTS  2048
#define DICT  32768
#define KTOP  64

// ---------------- WMMA types ----------------
typedef __attribute__((ext_vector_type(16))) __bf16 v16bf;
typedef __attribute__((ext_vector_type(8)))  float  v8f;

union Frag { v16bf v; uint4 q[2]; };

// bf16 round-to-nearest-even via integer math (no __bf16 arithmetic needed)
__device__ __forceinline__ unsigned short bf16_rne(float f) {
  unsigned u = __float_as_uint(f);
  u += 0x7FFFu + ((u >> 16) & 1u);
  return (unsigned short)(u >> 16);
}
__device__ __forceinline__ float bf16_to_f(unsigned short h) {
  return __uint_as_float(((unsigned)h) << 16);
}

// =====================================================================
// Kernel 1: encode GEMM  C[m,n] = sum_k x[m,k]*W_enc[n,k] + b_enc[n]
// bf16 hi/lo split (3 WMMA products) for ~fp32 accuracy at bf16 rate.
// Block tile 128x128, BK=32, 8 waves (2x4), wave tile 64x32.
// =====================================================================
#define BM 128
#define BN 128
#define BK 32
#define LDT 40   // padded LDS row stride (bf16 elements)

__global__ __launch_bounds__(256) void sae_encode_wmma(
    const float* __restrict__ x, const float* __restrict__ W,
    const float* __restrict__ b, float* __restrict__ C) {
  __shared__ __align__(16) unsigned short sAh[BM * LDT];
  __shared__ __align__(16) unsigned short sAl[BM * LDT];
  __shared__ __align__(16) unsigned short sBh[BN * LDT];
  __shared__ __align__(16) unsigned short sBl[BN * LDT];

  const int tid  = threadIdx.x;
  const int lane = tid & 31;
  const int wave = tid >> 5;
  const int half = lane >> 4;     // 0: lanes 0-15, 1: lanes 16-31
  const int l16  = lane & 15;
  const int wm   = wave >> 2;     // 0..1  (M direction)
  const int wn   = wave & 3;      // 0..3  (N direction)

  const size_t blockM = (size_t)blockIdx.x * BM;   // M fastest -> W_enc stripe L2 reuse
  const size_t blockN = (size_t)blockIdx.y * BN;

  v8f acc[4][2];
#pragma unroll
  for (int m = 0; m < 4; ++m)
#pragma unroll
    for (int n = 0; n < 2; ++n)
#pragma unroll
      for (int r = 0; r < 8; ++r) acc[m][n][r] = 0.0f;

  const int ldRow = tid >> 3;  // 0..31
  const int ldC4  = tid & 7;   // 0..7 (float4 column)

  for (int kt = 0; kt < ACTS; kt += BK) {
    __syncthreads();
    // ---- global fp32 -> split bf16 hi/lo -> LDS ----
#pragma unroll
    for (int s = 0; s < 4; ++s) {
      const int r = ldRow + s * 32;
      const float4 a4 = *(const float4*)&x[(blockM + r) * ACTS + kt + ldC4 * 4];
      const float4 w4 = *(const float4*)&W[(blockN + r) * ACTS + kt + ldC4 * 4];
      const float av[4] = {a4.x, a4.y, a4.z, a4.w};
      const float wv[4] = {w4.x, w4.y, w4.z, w4.w};
#pragma unroll
      for (int i = 0; i < 4; ++i) {
        unsigned short h = bf16_rne(av[i]);
        sAh[r * LDT + ldC4 * 4 + i] = h;
        sAl[r * LDT + ldC4 * 4 + i] = bf16_rne(av[i] - bf16_to_f(h));
        unsigned short g = bf16_rne(wv[i]);
        sBh[r * LDT + ldC4 * 4 + i] = g;
        sBl[r * LDT + ldC4 * 4 + i] = bf16_rne(wv[i] - bf16_to_f(g));
      }
    }
    __syncthreads();

    // ---- assemble WMMA fragments from LDS ----
    Frag ah[4], al[4], bh[2], bl[2];
#pragma unroll
    for (int m = 0; m < 4; ++m) {
      const int mr = wm * 64 + m * 16 + l16;
      const int kb = half * 8;                       // lanes>=16 start at K=8
      const unsigned short* p = &sAh[mr * LDT + kb];
      ah[m].q[0] = *(const uint4*)p;                 // K kb..kb+7
      ah[m].q[1] = *(const uint4*)(p + 16);          // K kb+16..kb+23
      const unsigned short* q = &sAl[mr * LDT + kb];
      al[m].q[0] = *(const uint4*)q;
      al[m].q[1] = *(const uint4*)(q + 16);
    }
#pragma unroll
    for (int n = 0; n < 2; ++n) {
      const int nr = wn * 32 + n * 16 + l16;
      const int kb = half * 8;                       // lanes>=16 start at K=8
      const unsigned short* p = &sBh[nr * LDT + kb];
      bh[n].q[0] = *(const uint4*)p;
      bh[n].q[1] = *(const uint4*)(p + 16);
      const unsigned short* q = &sBl[nr * LDT + kb];
      bl[n].q[0] = *(const uint4*)q;
      bl[n].q[1] = *(const uint4*)(q + 16);
    }

    // ---- 24 WMMAs: hi*hi + hi*lo + lo*hi ----
#pragma unroll
    for (int m = 0; m < 4; ++m)
#pragma unroll
      for (int n = 0; n < 2; ++n) {
        acc[m][n] = __builtin_amdgcn_wmma_f32_16x16x32_bf16(
            false, ah[m].v, false, bh[n].v, (short)0, acc[m][n], false, false);
        acc[m][n] = __builtin_amdgcn_wmma_f32_16x16x32_bf16(
            false, ah[m].v, false, bl[n].v, (short)0, acc[m][n], false, false);
        acc[m][n] = __builtin_amdgcn_wmma_f32_16x16x32_bf16(
            false, al[m].v, false, bh[n].v, (short)0, acc[m][n], false, false);
      }
  }

  // ---- epilogue: + b_enc, store fp32 C ----
#pragma unroll
  for (int n = 0; n < 2; ++n) {
    const size_t col = blockN + wn * 32 + n * 16 + l16;
    const float bv = b[col];
#pragma unroll
    for (int m = 0; m < 4; ++m) {
      const size_t rowb = blockM + wm * 64 + m * 16 + half * 8;
#pragma unroll
      for (int r = 0; r < 8; ++r)
        C[(rowb + r) * (size_t)DICT + col] = acc[m][n][r] + bv;
    }
  }
}

// =====================================================================
// Kernel 2: per-row exact top-64 (radix select on sortable keys) +
// faithful scatter acts[row, topIdx[rank]] = pre_acts[row, rank<64].
// Operates in place on the acts region of d_out (pre_acts written there).
// =====================================================================
__device__ __forceinline__ unsigned keyOf(float f) {
  unsigned u = __float_as_uint(f);
  return (u & 0x80000000u) ? ~u : (u | 0x80000000u);  // larger float -> larger key
}

#define MAXCAND 160

__global__ __launch_bounds__(256) void sae_topk_scatter(
    float* __restrict__ acts, int* __restrict__ wsIdx, float* __restrict__ wsVal) {
  const int row = blockIdx.x;
  float* prow = acts + (size_t)row * DICT;
  const int tid = threadIdx.x;

  __shared__ unsigned hist[2048];
  __shared__ float firstVals[KTOP];
  __shared__ int selHi, selMid, selLo;
  __shared__ unsigned cntAboveHi, cntAboveMid;
  __shared__ int nCand;
  __shared__ unsigned candKey[MAXCAND];
  __shared__ int candIdx[MAXCAND];
  __shared__ int topIdxByRank[KTOP];

  // ---- pass A: top-11-bit histogram; capture first-64 values ----
  for (int i = tid; i < 2048; i += 256) hist[i] = 0;
  if (tid < KTOP) firstVals[tid] = prow[tid];
  __syncthreads();
  for (int i = tid; i < DICT; i += 256) atomicAdd(&hist[keyOf(prow[i]) >> 21], 1u);
  __syncthreads();
  if (tid == 0) {
    unsigned c = 0; int bk = 2047;
    for (; bk >= 0; --bk) { if (c + hist[bk] >= (unsigned)KTOP) break; c += hist[bk]; }
    selHi = bk; cntAboveHi = c;
  }
  __syncthreads();
  const unsigned sh = (unsigned)selHi;

  // ---- pass B: next 11 bits within selected bucket ----
  for (int i = tid; i < 2048; i += 256) hist[i] = 0;
  __syncthreads();
  for (int i = tid; i < DICT; i += 256) {
    unsigned k = keyOf(prow[i]);
    if ((k >> 21) == sh) atomicAdd(&hist[(k >> 10) & 2047u], 1u);
  }
  __syncthreads();
  if (tid == 0) {
    unsigned c = cntAboveHi; int bk = 2047;
    for (; bk >= 0; --bk) { if (c + hist[bk] >= (unsigned)KTOP) break; c += hist[bk]; }
    selMid = bk; cntAboveMid = c;
  }
  __syncthreads();
  const unsigned hiPrefix = (sh << 11) | (unsigned)selMid;   // == key>>10

  // ---- pass C: low 10 bits -> exact 64th-largest key ----
  for (int i = tid; i < 1024; i += 256) hist[i] = 0;
  __syncthreads();
  for (int i = tid; i < DICT; i += 256) {
    unsigned k = keyOf(prow[i]);
    if ((k >> 10) == hiPrefix) atomicAdd(&hist[k & 1023u], 1u);
  }
  __syncthreads();
  if (tid == 0) {
    unsigned c = cntAboveMid; int bk = 1023;
    for (; bk >= 0; --bk) { if (c + hist[bk] >= (unsigned)KTOP) break; c += hist[bk]; }
    selLo = bk; nCand = 0;
  }
  __syncthreads();
  const unsigned T = (hiPrefix << 10) | (unsigned)selLo;

  // ---- pass D: collect candidates key >= T, rank with top_k tie order ----
  for (int i = tid; i < DICT; i += 256) {
    unsigned k = keyOf(prow[i]);
    if (k >= T) {
      int s = atomicAdd(&nCand, 1);
      if (s < MAXCAND) { candKey[s] = k; candIdx[s] = i; }
    }
  }
  __syncthreads();
  const int nc = (nCand < MAXCAND) ? nCand : MAXCAND;
  for (int c = tid; c < nc; c += 256) {
    const unsigned kc = candKey[c];
    const int ic = candIdx[c];
    int r = 0;
    for (int j = 0; j < nc; ++j) {
      const unsigned kj = candKey[j];
      r += (kj > kc || (kj == kc && candIdx[j] < ic)) ? 1 : 0;
    }
    if (r < KTOP) topIdxByRank[r] = ic;
  }
  __syncthreads();

  // ---- pass E: zero row, then scatter first-64 values to topk positions ----
  for (int i = tid; i < DICT; i += 256) prow[i] = 0.0f;
  __threadfence();
  __syncthreads();
  if (tid < KTOP) {
    const int ix = topIdxByRank[tid];
    const float v = firstVals[tid];
    prow[ix] = v;
    wsIdx[row * KTOP + tid] = ix;
    wsVal[row * KTOP + tid] = v;
  }
}

// =====================================================================
// Kernel 3: transpose W_dec [2048][32768] -> Wt [32768][2048] (fp32)
// so decode gathers contiguous 8KB rows instead of 128KB-strided columns.
// =====================================================================
__global__ __launch_bounds__(256) void sae_transpose_wdec(
    const float* __restrict__ src, float* __restrict__ dst) {
  __shared__ float tile[64][65];
  const int bx = blockIdx.x;           // dict tiles (32768/64)
  const int by = blockIdx.y;           // acts tiles (2048/64)
  const int tx = threadIdx.x & 63;
  const int ty = threadIdx.x >> 6;     // 0..3
  for (int r = ty; r < 64; r += 4)
    tile[r][tx] = src[((size_t)by * 64 + r) * DICT + (size_t)bx * 64 + tx];
  __syncthreads();
  for (int r = ty; r < 64; r += 4)
    dst[((size_t)bx * 64 + r) * ACTS + (size_t)by * 64 + tx] = tile[tx][r];
}

// =====================================================================
// Kernel 4: sparse decode  recon[n, :] = sum_j val[n,j] * Wt[idx[n,j], :]
// one block per token; each thread owns 8 contiguous outputs.
// =====================================================================
__global__ __launch_bounds__(256) void sae_decode_sparse(
    const float* __restrict__ Wt, const int* __restrict__ wsIdx,
    const float* __restrict__ wsVal, float* __restrict__ recon) {
  const int row = blockIdx.x;
  const int tid = threadIdx.x;
  __shared__ int sIdx[KTOP];
  __shared__ float sVal[KTOP];
  if (tid < KTOP) { sIdx[tid] = wsIdx[row * KTOP + tid]; sVal[tid] = wsVal[row * KTOP + tid]; }
  __syncthreads();

  float acc[8];
#pragma unroll
  for (int i = 0; i < 8; ++i) acc[i] = 0.0f;
  const int a0 = tid * 8;

  for (int j = 0; j < KTOP; ++j) {
    const float v = sVal[j];
    const float4* p = (const float4*)&Wt[(size_t)sIdx[j] * ACTS + a0];
    const float4 w0 = p[0], w1 = p[1];
    acc[0] += v * w0.x; acc[1] += v * w0.y; acc[2] += v * w0.z; acc[3] += v * w0.w;
    acc[4] += v * w1.x; acc[5] += v * w1.y; acc[6] += v * w1.z; acc[7] += v * w1.w;
  }
  float4* out = (float4*)&recon[(size_t)row * ACTS + a0];
  out[0] = make_float4(acc[0], acc[1], acc[2], acc[3]);
  out[1] = make_float4(acc[4], acc[5], acc[6], acc[7]);
}

// =====================================================================
extern "C" void kernel_launch(void* const* d_in, const int* in_sizes, int n_in,
                              void* d_out, int out_size, void* d_ws, size_t ws_size,
                              hipStream_t stream) {
  const float* x     = (const float*)d_in[0];   // [4096, 2048]
  const float* W_enc = (const float*)d_in[1];   // [32768, 2048]
  const float* b_enc = (const float*)d_in[2];   // [32768]
  const float* W_dec = (const float*)d_in[3];   // [2048, 32768]

  float* recon = (float*)d_out;                       // [4096, 2048]
  float* acts  = recon + (size_t)TOK * ACTS;          // [4096, 32768]

  int*   wsIdx = (int*)d_ws;                                    // 1 MB
  float* wsVal = (float*)((char*)d_ws + (1u << 20));            // 1 MB
  float* Wt    = (float*)((char*)d_ws + (2u << 20));            // 256 MB transposed W_dec

  // 1) encode GEMM (WMMA bf16x3): pre_acts -> acts region of d_out
  sae_encode_wmma<<<dim3(TOK / BM, DICT / BN), 256, 0, stream>>>(x, W_enc, b_enc, acts);
  // 2) transpose W_dec for contiguous decode gathers (independent of 1)
  sae_transpose_wdec<<<dim3(DICT / 64, ACTS / 64), 256, 0, stream>>>(W_dec, Wt);
  // 3) exact per-row top-64 + faithful scatter (in place in acts)
  sae_topk_scatter<<<TOK, 256, 0, stream>>>(acts, wsIdx, wsVal);
  // 4) sparse decode -> recon
  sae_decode_sparse<<<TOK, 256, 0, stream>>>(Wt, wsIdx, wsVal, recon);
}
